// AttentionBlock_4294967296147
// MI455X (gfx1250) — compile-verified, matrix operand order checked by reading, not run on hardware
//
#include <hip/hip_runtime.h>

typedef __attribute__((ext_vector_type(16))) __bf16 v16bf;
typedef __attribute__((ext_vector_type(8)))  __bf16 v8bf;
typedef __attribute__((ext_vector_type(8)))  float  v8f;
typedef __attribute__((ext_vector_type(4)))  unsigned int v4u;
typedef __attribute__((ext_vector_type(8)))  int v8i;
typedef __attribute__((ext_vector_type(4)))  int v4i;

#define BDIM 256
static constexpr int Bc = 4, Sc = 2048, Ec = 1024, Hc = 16, Dc = 64;
static constexpr int Mrows = Bc * Sc;          // 8192 token rows
static constexpr float NEGBIG = -3.0e38f;

// Load a 16x32 bf16 WMMA A/B fragment from a row-major tile.
// Lane L: row = L%16, K elements (L/16)*8 .. +7 and +16 .. +23 (ISA 7.12.2).
__device__ __forceinline__ v16bf ldfrag(const __bf16* p) {
  union { v16bf v; v8bf h[2]; } u;
  u.h[0] = *(const v8bf*)(p);
  u.h[1] = *(const v8bf*)(p + 16);
  return u.v;
}

__device__ __forceinline__ v8f wmma_bf16(v16bf a, v16bf b, v8f c) {
  return __builtin_amdgcn_wmma_f32_16x16x32_bf16(false, a, false, b, (short)0, c,
                                                 false, false);
}

#if __has_builtin(__builtin_amdgcn_tensor_load_to_lds)
#define USE_TDM 1
// Issue a TDM 2D tile load (bf16 elements) described by a hand-packed D#.
// tile: tile_k x tile_rows elements; tensor: tensor_k x tensor_rows with row
// stride stride_k (elements). Tracked by TENSORcnt.
__device__ __forceinline__ void tdm_load_2d(unsigned lds_addr, const void* gaddr,
                                            unsigned tile_k, unsigned tile_rows,
                                            unsigned tensor_k, unsigned tensor_rows,
                                            unsigned stride_k) {
  unsigned long long ga = (unsigned long long)(uintptr_t)gaddr;
  v4u g0;
  g0[0] = 1u;                                     // count=1, user descriptor
  g0[1] = lds_addr;                               // LDS byte address
  g0[2] = (unsigned)(ga & 0xffffffffu);           // global_addr[31:0]
  g0[3] = (unsigned)((ga >> 32) & 0x01ffffffu)    // global_addr[56:32]
          | (2u << 30);                           // type = 2 ("image")
  v8i g1;
  g1[0] = (int)(1u << 16);                        // wg_mask=0, data_size=1 (2B)
  g1[1] = (int)((tensor_k & 0xffffu) << 16);      // tensor_dim0[15:0]
  g1[2] = (int)((tensor_k >> 16) | ((tensor_rows & 0xffffu) << 16));
  g1[3] = (int)((tensor_rows >> 16) | (tile_k << 16));   // tile_dim0
  g1[4] = (int)(tile_rows & 0xffffu);             // tile_dim1; tile_dim2=0
  g1[5] = (int)stride_k;                          // tensor_dim0_stride[31:0]
  g1[6] = 0;                                      // stride hi / dim1_stride lo
  g1[7] = 0;
  v4i z4 = {0, 0, 0, 0};
  v8i z8 = {0, 0, 0, 0, 0, 0, 0, 0};
  __builtin_amdgcn_tensor_load_to_lds(g0, g1, z4, z4, z8, 0);
}
#else
#define USE_TDM 0
#endif

// Block-cooperative GEMM core: C[128x64] per block, 8 waves x (16x64 tiles).
// acc[f] accumulates the wave's 16x16 tile for n-chunk f.
// sA: 2 x (128x32) bf16, sB: 2 x (64x32) bf16 double-buffered LDS tiles.
__device__ __forceinline__ void gemm_block(const __bf16* __restrict__ Amat,
                                           const __bf16* __restrict__ Bmat,
                                           int m0, int n0, int rowsB,
                                           __bf16* sA, __bf16* sB, v8f acc[4]) {
  int wid = threadIdx.x >> 5, lane = threadIdx.x & 31;
  int rsel = lane & 15, half = lane >> 4;
#if USE_TDM
  const unsigned tileA = 128 * 32, tileB = 64 * 32;   // elements
  unsigned ldsA = (unsigned)(size_t)(void*)sA;
  unsigned ldsB = (unsigned)(size_t)(void*)sB;
  if (wid == 0) {
    tdm_load_2d(ldsA, Amat + (size_t)m0 * Ec, 32, 128, Ec, Mrows, Ec);
    tdm_load_2d(ldsB, Bmat + (size_t)n0 * Ec, 32, 64, Ec, rowsB, Ec);
    __builtin_amdgcn_s_wait_tensorcnt(0);
  }
  __syncthreads();
  for (int k0 = 0; k0 < Ec; k0 += 32) {
    int cur = (k0 >> 5) & 1, nxt = cur ^ 1;
    bool more = (k0 + 32) < Ec;
    if (wid == 0 && more) {   // prefetch next k-tile while computing this one
      tdm_load_2d(ldsA + nxt * tileA * 2, Amat + (size_t)m0 * Ec + k0 + 32,
                  32, 128, Ec, Mrows, Ec);
      tdm_load_2d(ldsB + nxt * tileB * 2, Bmat + (size_t)n0 * Ec + k0 + 32,
                  32, 64, Ec, rowsB, Ec);
    }
    // Batch all fragment loads first so the four WMMAs can issue back-to-back
    // behind partial s_wait_dscnt waits.
    const __bf16* aP = sA + cur * tileA + (wid * 16 + rsel) * 32 + half * 8;
    v16bf a = ldfrag(aP);
    v16bf b[4];
#pragma unroll
    for (int f = 0; f < 4; f++)
      b[f] = ldfrag(sB + cur * tileB + (f * 16 + rsel) * 32 + half * 8);
#pragma unroll
    for (int f = 0; f < 4; f++)
      acc[f] = wmma_bf16(a, b[f], acc[f]);
    if (wid == 0 && more) __builtin_amdgcn_s_wait_tensorcnt(0);
    __syncthreads();
  }
#else
  // Fallback: direct per-wave global fragment loads.
  (void)sA; (void)sB; (void)rowsB;
  const __bf16* aBase = Amat + (size_t)(m0 + wid * 16 + rsel) * Ec + half * 8;
  const __bf16* bBase = Bmat + (size_t)(n0 + rsel) * Ec + half * 8;
  for (int k0 = 0; k0 < Ec; k0 += 32) {
    __builtin_prefetch(aBase + k0 + 64, 0, 0);
    v16bf a = ldfrag(aBase + k0);
    v16bf b[4];
#pragma unroll
    for (int f = 0; f < 4; f++)
      b[f] = ldfrag(bBase + (size_t)f * 16 * Ec + k0);
#pragma unroll
    for (int f = 0; f < 4; f++)
      acc[f] = wmma_bf16(a, b[f], acc[f]);
  }
#endif
}

// ---------------- Kernel 0: f32 -> bf16 weight cast ----------------
__global__ void cast_bf16_kernel(const float* __restrict__ in,
                                 __bf16* __restrict__ out, int n) {
  int i = blockIdx.x * blockDim.x + threadIdx.x;
  if (i < n) out[i] = (__bf16)in[i];
}

// ---------------- Kernel 1: LayerNorm + bf16 cast ----------------
__global__ void __launch_bounds__(BDIM)
ln_cast_kernel(const float* __restrict__ x, const float* __restrict__ w,
               __bf16* __restrict__ xn) {
  int row = blockIdx.x;
  const float* xr = x + (size_t)row * Ec;
  float v[4];
  float s = 0.f, ss = 0.f;
#pragma unroll
  for (int i = 0; i < 4; i++) {
    v[i] = xr[threadIdx.x + i * BDIM];
    s += v[i];
    ss += v[i] * v[i];
  }
#pragma unroll
  for (int m = 16; m >= 1; m >>= 1) {
    s  += __shfl_xor(s, m, 32);
    ss += __shfl_xor(ss, m, 32);
  }
  __shared__ float reds[BDIM / 32], redss[BDIM / 32];
  int wid = threadIdx.x >> 5;
  if ((threadIdx.x & 31) == 0) { reds[wid] = s; redss[wid] = ss; }
  __syncthreads();
  float tot = 0.f, tots = 0.f;
#pragma unroll
  for (int i = 0; i < BDIM / 32; i++) { tot += reds[i]; tots += redss[i]; }
  float mu  = tot / (float)Ec;
  float var = tots / (float)Ec - mu * mu;
  float inv = rsqrtf(var + 1e-5f);
  __bf16* xo = xn + (size_t)row * Ec;
#pragma unroll
  for (int i = 0; i < 4; i++) {
    int c = threadIdx.x + i * BDIM;
    xo[c] = (__bf16)((v[i] - mu) * inv * w[c]);
  }
}

// ---------------- Kernel 2: QKV GEMM (block 128x64, TDM-staged) ----------------
__global__ void __launch_bounds__(BDIM)
qkv_gemm_kernel(const __bf16* __restrict__ xn, const __bf16* __restrict__ wq,
                __bf16* __restrict__ Qh, __bf16* __restrict__ Kh,
                __bf16* __restrict__ Vt) {
  __shared__ __bf16 sA[2 * 128 * 32];
  __shared__ __bf16 sB[2 * 64 * 32];
  int wid = threadIdx.x >> 5, lane = threadIdx.x & 31;
  int rsel = lane & 15, half = lane >> 4;
  int bm = blockIdx.x & 63, bn = blockIdx.x >> 6;   // 64 x 48 blocks
  int m0 = bm * 128, n0 = bn * 64;

  v8f acc[4] = {};
  gemm_block(xn, wq, m0, n0, 3 * Ec, sA, sB, acc);

  int rowbase = m0 + wid * 16 + half * 8;
#pragma unroll
  for (int f = 0; f < 4; f++) {
    int gn  = n0 + f * 16 + rsel;   // 0..3071
    int sec = gn >> 10;             // 0=Q 1=K 2=V
    int e   = gn & 1023;
    int h   = e >> 6, d = e & 63;
#pragma unroll
    for (int r = 0; r < 8; r++) {
      int grow = rowbase + r;
      int b = grow >> 11, srow = grow & (Sc - 1);
      float val = acc[f][r];
      if (sec == 0)
        Qh[((size_t)(b * Hc + h) * Sc + srow) * Dc + d] = (__bf16)(val * 0.125f);
      else if (sec == 1)
        Kh[((size_t)(b * Hc + h) * Sc + srow) * Dc + d] = (__bf16)val;
      else
        Vt[((size_t)(b * Hc + h) * Dc + d) * Sc + srow] = (__bf16)val;
    }
  }
}

// ---------------- Kernel 3: causal flash attention with ALiBi bias ----------------
__global__ void __launch_bounds__(BDIM)
flash_kernel(const __bf16* __restrict__ Qh, const __bf16* __restrict__ Kh,
             const __bf16* __restrict__ Vt, const float* __restrict__ scaler,
             __bf16* __restrict__ AO) {
  __shared__ __bf16 lds_p[BDIM / 32][16][32];
  int wid = threadIdx.x >> 5, lane = threadIdx.x & 31;
  int rsel = lane & 15, half = lane >> 4, rowoff = half * 8;

  int tile = blockIdx.x * (BDIM / 32) + wid;  // 8192 tiles = 64 bh * 128 qtiles
  int qi = tile & 127, bh = tile >> 7;
  int b = bh >> 4, h = bh & 15;
  int sq0 = qi * 16;

  float slope = log1pf(__expf(100.0f * scaler[0]));

  const __bf16* qbase = Qh + ((size_t)bh * Sc + sq0 + rsel) * Dc + half * 8;
  v16bf qf0 = ldfrag(qbase);
  v16bf qf1 = ldfrag(qbase + 32);
  const __bf16* kbase = Kh + (size_t)bh * Sc * Dc;
  const __bf16* vbase = Vt + (size_t)bh * Dc * Sc;

  float mi[8], li[8];
#pragma unroll
  for (int r = 0; r < 8; r++) { mi[r] = NEGBIG; li[r] = 0.f; }
  v8f o[4] = {};

  for (int j0 = 0; j0 <= sq0 + 15; j0 += 32) {
    // Batch K fragment loads for both 16-key chunks, then both score WMMAs.
    v16bf kf[2][2];
#pragma unroll
    for (int jj = 0; jj < 2; jj++) {
      const __bf16* kp = kbase + (size_t)(j0 + jj * 16 + rsel) * Dc + half * 8;
      kf[jj][0] = ldfrag(kp);
      kf[jj][1] = ldfrag(kp + 32);
    }
    float p[2][8], rm[8];
#pragma unroll
    for (int r = 0; r < 8; r++) rm[r] = NEGBIG;
#pragma unroll
    for (int jj = 0; jj < 2; jj++) {
      int jg = j0 + jj * 16 + rsel;                 // key (column) this lane owns
      v8f sf = {};
      sf = wmma_bf16(qf0, kf[jj][0], sf);           // Q already holds 1/sqrt(D)
      sf = wmma_bf16(qf1, kf[jj][1], sf);
#pragma unroll
      for (int r = 0; r < 8; r++) {
        int ig = sq0 + rowoff + r;
        float v = sf[r];
        v = (jg <= ig) ? (v + slope * (float)(jg - ig)) : NEGBIG;
        p[jj][r] = v;
        rm[r] = fmaxf(rm[r], v);
      }
    }
#pragma unroll
    for (int r = 0; r < 8; r++) {
      float v = rm[r];
      v = fmaxf(v, __shfl_xor(v, 1, 32));
      v = fmaxf(v, __shfl_xor(v, 2, 32));
      v = fmaxf(v, __shfl_xor(v, 4, 32));
      v = fmaxf(v, __shfl_xor(v, 8, 32));
      rm[r] = v;
    }
    float corr[8], psum[8];
#pragma unroll
    for (int r = 0; r < 8; r++) {
      float mnew = fmaxf(mi[r], rm[r]);
      corr[r] = __expf(mi[r] - mnew);
      mi[r] = mnew;
      float e0 = __expf(p[0][r] - mnew);
      float e1 = __expf(p[1][r] - mnew);
      p[0][r] = e0; p[1][r] = e1;
      psum[r] = e0 + e1;
    }
#pragma unroll
    for (int r = 0; r < 8; r++) {
      float v = psum[r];
      v += __shfl_xor(v, 1, 32);
      v += __shfl_xor(v, 2, 32);
      v += __shfl_xor(v, 4, 32);
      v += __shfl_xor(v, 8, 32);
      li[r] = li[r] * corr[r] + v;
    }
#pragma unroll
    for (int c = 0; c < 4; c++)
#pragma unroll
      for (int r = 0; r < 8; r++) o[c][r] *= corr[r];

    // C-layout -> A-layout relayout of P through per-wave LDS staging
#pragma unroll
    for (int jj = 0; jj < 2; jj++)
#pragma unroll
      for (int r = 0; r < 8; r++)
        lds_p[wid][rowoff + r][jj * 16 + rsel] = (__bf16)p[jj][r];
    asm volatile("s_wait_dscnt 0" ::: "memory");   // wave-local LDS RAW fence
    v16bf pf = ldfrag(&lds_p[wid][rsel][half * 8]);

    // Batch all V fragments, then issue the four P·V WMMAs back-to-back.
    v16bf vf[4];
#pragma unroll
    for (int c = 0; c < 4; c++)
      vf[c] = ldfrag(vbase + (size_t)(c * 16 + rsel) * Sc + j0 + half * 8);
#pragma unroll
    for (int c = 0; c < 4; c++)
      o[c] = wmma_bf16(pf, vf[c], o[c]);
  }

#pragma unroll
  for (int c = 0; c < 4; c++)
#pragma unroll
    for (int r = 0; r < 8; r++) {
      int srow = sq0 + rowoff + r;
      AO[(size_t)(b * Sc + srow) * Ec + h * 64 + c * 16 + rsel] =
          (__bf16)(o[c][r] / li[r]);
    }
}

// ---------------- Kernel 4: output projection + residual (TDM-staged) ----------------
__global__ void __launch_bounds__(BDIM)
oproj_kernel(const __bf16* __restrict__ AO, const __bf16* __restrict__ Wo,
             const float* __restrict__ x, float* __restrict__ out) {
  __shared__ __bf16 sA[2 * 128 * 32];
  __shared__ __bf16 sB[2 * 64 * 32];
  int wid = threadIdx.x >> 5, lane = threadIdx.x & 31;
  int rsel = lane & 15, half = lane >> 4;
  int bm = blockIdx.x & 63, bn = blockIdx.x >> 6;   // 64 x 16 blocks
  int m0 = bm * 128, n0 = bn * 64;

  v8f acc[4] = {};
  gemm_block(AO, Wo, m0, n0, Ec, sA, sB, acc);

#pragma unroll
  for (int f = 0; f < 4; f++) {
    int col = n0 + f * 16 + rsel;
#pragma unroll
    for (int r = 0; r < 8; r++) {
      size_t idx = (size_t)(m0 + wid * 16 + half * 8 + r) * Ec + col;
      out[idx] = acc[f][r] + x[idx];
    }
  }
}

// ---------------- Host launcher ----------------
extern "C" void kernel_launch(void* const* d_in, const int* in_sizes, int n_in,
                              void* d_out, int out_size, void* d_ws, size_t ws_size,
                              hipStream_t stream) {
  (void)in_sizes; (void)n_in; (void)out_size; (void)ws_size;
  const float* x      = (const float*)d_in[0];
  const float* Wqkv   = (const float*)d_in[1];
  const float* Wo     = (const float*)d_in[2];
  const float* lnw    = (const float*)d_in[3];
  const float* scaler = (const float*)d_in[4];
  float* out = (float*)d_out;

  char* ws = (char*)d_ws;
  const size_t nWqkv = (size_t)3 * Ec * Ec;
  const size_t nWo   = (size_t)Ec * Ec;
  const size_t nTok  = (size_t)Mrows * Ec;
  __bf16* wqkv_bf = (__bf16*)(ws);
  __bf16* wo_bf   = (__bf16*)(ws + 2 * nWqkv);
  __bf16* xn      = (__bf16*)(ws + 2 * (nWqkv + nWo));
  __bf16* Qh      = (__bf16*)(ws + 2 * (nWqkv + nWo + nTok));
  __bf16* Kh      = (__bf16*)(ws + 2 * (nWqkv + nWo + 2 * nTok));
  __bf16* Vt      = (__bf16*)(ws + 2 * (nWqkv + nWo + 3 * nTok));
  __bf16* AO      = (__bf16*)(ws + 2 * (nWqkv + nWo + 4 * nTok));

  cast_bf16_kernel<<<(int)((nWqkv + BDIM - 1) / BDIM), BDIM, 0, stream>>>(
      Wqkv, wqkv_bf, (int)nWqkv);
  cast_bf16_kernel<<<(int)((nWo + BDIM - 1) / BDIM), BDIM, 0, stream>>>(
      Wo, wo_bf, (int)nWo);
  ln_cast_kernel<<<Mrows, BDIM, 0, stream>>>(x, lnw, xn);
  // 64 m-blocks * 48 n-blocks
  qkv_gemm_kernel<<<64 * 48, BDIM, 0, stream>>>(xn, wqkv_bf, Qh, Kh, Vt);
  // 64 bh * 128 q-tiles = 8192 wave tiles
  flash_kernel<<<8192 / (BDIM / 32), BDIM, 0, stream>>>(Qh, Kh, Vt, scaler, AO);
  // 64 m-blocks * 16 n-blocks
  oproj_kernel<<<64 * 16, BDIM, 0, stream>>>(AO, wo_bf, x, out);
}
